// CausalSelfAttention_2448131359201
// MI455X (gfx1250) — matrix operand order repaired and checked by reading, not compiled, not run on hardware
//
#include <hip/hip_runtime.h>
#include <math.h>

// Causal self-attention forward for MI455X (gfx1250).
// wave32 + WMMA bf16, async global->LDS (ASYNCcnt) for K tiles,
// Tensor Data Mover (TENSORcnt) for strided V^T tiles, double-buffered LDS.
// B=2, S=2048, D=1024, H=16, head_dim=64.

#define S_LEN   2048
#define NH      16
#define HD      64
#define DMODEL  1024
#define BATCH   2
#define BHCNT   (BATCH * NH)

typedef __attribute__((ext_vector_type(16))) __bf16 v16bf;
typedef __attribute__((ext_vector_type(8)))  __bf16 v8bf;
typedef __attribute__((ext_vector_type(4)))  __bf16 v4bf;
typedef __attribute__((ext_vector_type(8)))  float  v8f;
typedef __attribute__((ext_vector_type(4)))  unsigned u32x4;
typedef __attribute__((ext_vector_type(8)))  unsigned u32x8;

union FragU { v16bf v; uint4 q[2]; };

// A-matrix 16x32 bf16 fragment (ISA 7.12.2): lane l holds row (l&15);
// half sel=(l>>4): halves 0..7 = K[sel*8..+7], halves 8..15 = K[16+sel*8..+7].
__device__ __forceinline__ v16bf load_frag_a(const __bf16* row_ptr, int sel) {
  FragU u;
  u.q[0] = *(const uint4*)(row_ptr + sel * 8);
  u.q[1] = *(const uint4*)(row_ptr + 16 + sel * 8);
  return u.v;
}
// B-matrix 32x16 bf16 fragment: lane l holds column (l&15);
// half sel=(l>>4): halves 0..7 = K[sel*8..+7], halves 8..15 = K[16+sel*8..+7].
__device__ __forceinline__ v16bf load_frag_b(const __bf16* col_ptr, int sel) {
  FragU u;
  u.q[0] = *(const uint4*)(col_ptr + sel * 8);
  u.q[1] = *(const uint4*)(col_ptr + 16 + sel * 8);
  return u.v;
}

// One 16-byte per-lane async copy global -> LDS (ASYNCcnt-tracked).
__device__ __forceinline__ void async_cp16(const __bf16* gsrc, __bf16* ldst) {
  unsigned l = (unsigned)(unsigned long long)ldst;          // LDS byte offset
  unsigned long long g = (unsigned long long)gsrc;
  asm volatile("global_load_async_to_lds_b128 %0, %1, off"
               :: "v"(l), "v"(g) : "memory");
}

// Tensor Data Mover: load a 2D tile (tile_dim0=32 elems x tile_dim1=64 rows,
// 2-byte elements, row stride = S_LEN elems) from global into contiguous LDS.
// D# built per ISA ch.8: group0 = {flags, lds_addr, global_addr, type},
// group1 = {data_size, tensor dims, tile dims, dim0 stride}. TENSORcnt-tracked.
__device__ __forceinline__ void tdm_load_vtile(const __bf16* gsrc, __bf16* ldst) {
  unsigned lds  = (unsigned)__builtin_amdgcn_readfirstlane(
                      (int)(unsigned)(unsigned long long)ldst);
  unsigned long long ga = (unsigned long long)gsrc;
  unsigned galo = (unsigned)__builtin_amdgcn_readfirstlane((int)(unsigned)ga);
  unsigned gahi = (unsigned)__builtin_amdgcn_readfirstlane(
                      (int)((unsigned)(ga >> 32) & 0x1FFFFFFu));
  u32x4 g0;
  g0[0] = 1u;                      // count=1 (valid user descriptor)
  g0[1] = lds;                     // lds_addr (bytes)
  g0[2] = galo;                    // global_addr[31:0]
  g0[3] = gahi | (2u << 30);       // global_addr[56:32] | type=2 ("image")
  u32x8 g1;
  g1[0] = 1u << 16;                // workgroup_mask=0, data_size=1 (2 bytes)
  g1[1] = 32u << 16;               // tensor_dim0 = 32 (lo16 at bits 63:48)
  g1[2] = (32u >> 16) | (64u << 16);   // tensor_dim0 hi | tensor_dim1=64 lo16
  g1[3] = (64u >> 16) | (32u << 16);   // tensor_dim1 hi | tile_dim0=32
  g1[4] = 64u;                     // tile_dim1 = 64, tile_dim2 = 0
  g1[5] = (unsigned)S_LEN;         // tensor_dim0_stride = 2048 elems
  g1[6] = 0u;                      // stride hi / tensor_dim1_stride lo
  g1[7] = 0u;
  asm volatile("tensor_load_to_lds %0, %1" :: "s"(g0), "s"(g1) : "memory");
}

// Reductions across the 16 lanes of a wave32 half (matches C-fragment rows).
__device__ __forceinline__ float half_red_max(float v) {
  v = fmaxf(v, __shfl_xor(v, 1, 32));
  v = fmaxf(v, __shfl_xor(v, 2, 32));
  v = fmaxf(v, __shfl_xor(v, 4, 32));
  v = fmaxf(v, __shfl_xor(v, 8, 32));
  return v;
}
__device__ __forceinline__ float half_red_sum(float v) {
  v += __shfl_xor(v, 1, 32);
  v += __shfl_xor(v, 2, 32);
  v += __shfl_xor(v, 4, 32);
  v += __shfl_xor(v, 8, 32);
  return v;
}

// ---------------- prep: fp32 -> bf16 ----------------
__global__ __launch_bounds__(256) void cvt_x_bf16(const float4* __restrict__ x,
                                                  __bf16* __restrict__ xb) {
  size_t i = (size_t)blockIdx.x * 256 + threadIdx.x;
  float4 f = x[i];
  v4bf o;
  o.x = (__bf16)f.x; o.y = (__bf16)f.y; o.z = (__bf16)f.z; o.w = (__bf16)f.w;
  *(v4bf*)(xb + i * 4) = o;
}

// w_qkv [1024,3072] fp32 -> wt [3072,1024] bf16 (transpose: B cols contiguous)
__global__ __launch_bounds__(256) void cvt_wt_bf16(const float* __restrict__ w,
                                                   __bf16* __restrict__ wt) {
  size_t i = (size_t)blockIdx.x * 256 + threadIdx.x;
  int k = (int)(i / 3072);
  int n = (int)(i - (size_t)k * 3072);
  wt[(size_t)n * DMODEL + k] = (__bf16)w[i];
}

// ---------------- QKV projection GEMM ----------------
// C[4096,3072] = Xb[4096,1024] @ Wt^T ; 8 waves/block, wave tile 32x64.
__global__ __launch_bounds__(256) void qkv_gemm(const __bf16* __restrict__ A,
                                                const __bf16* __restrict__ Bt,
                                                __bf16* __restrict__ qb,
                                                __bf16* __restrict__ kb,
                                                __bf16* __restrict__ vt) {
  const int lane = threadIdx.x & 31;
  const int wave = threadIdx.x >> 5;
  const int sel  = lane >> 4;
  const int ln   = lane & 15;
  const int wr   = wave >> 1;                    // 0..3
  const int wc   = wave & 1;                     // 0..1
  const int m0   = blockIdx.y * 128 + wr * 32;
  const int n0   = blockIdx.x * 128 + wc * 64;

  v8f acc[2][4] = {};
  const __bf16* arow0 = A  + (size_t)(m0 + ln) * DMODEL;
  const __bf16* arow1 = A  + (size_t)(m0 + 16 + ln) * DMODEL;
  const __bf16* brow0 = Bt + (size_t)(n0 + ln) * DMODEL;
  const __bf16* brow1 = Bt + (size_t)(n0 + 16 + ln) * DMODEL;
  const __bf16* brow2 = Bt + (size_t)(n0 + 32 + ln) * DMODEL;
  const __bf16* brow3 = Bt + (size_t)(n0 + 48 + ln) * DMODEL;

#pragma unroll 2
  for (int k = 0; k < DMODEL; k += 32) {
    __builtin_prefetch(arow0 + k + 128, 0, 0);
    __builtin_prefetch(brow0 + k + 128, 0, 0);
    v16bf a0 = load_frag_a(arow0 + k, sel);
    v16bf a1 = load_frag_a(arow1 + k, sel);
    v16bf b0 = load_frag_b(brow0 + k, sel);
    v16bf b1 = load_frag_b(brow1 + k, sel);
    v16bf b2 = load_frag_b(brow2 + k, sel);
    v16bf b3 = load_frag_b(brow3 + k, sel);
    acc[0][0] = __builtin_amdgcn_wmma_f32_16x16x32_bf16(false, a0, false, b0, (short)0, acc[0][0], false, false);
    acc[0][1] = __builtin_amdgcn_wmma_f32_16x16x32_bf16(false, a0, false, b1, (short)0, acc[0][1], false, false);
    acc[0][2] = __builtin_amdgcn_wmma_f32_16x16x32_bf16(false, a0, false, b2, (short)0, acc[0][2], false, false);
    acc[0][3] = __builtin_amdgcn_wmma_f32_16x16x32_bf16(false, a0, false, b3, (short)0, acc[0][3], false, false);
    acc[1][0] = __builtin_amdgcn_wmma_f32_16x16x32_bf16(false, a1, false, b0, (short)0, acc[1][0], false, false);
    acc[1][1] = __builtin_amdgcn_wmma_f32_16x16x32_bf16(false, a1, false, b1, (short)0, acc[1][1], false, false);
    acc[1][2] = __builtin_amdgcn_wmma_f32_16x16x32_bf16(false, a1, false, b2, (short)0, acc[1][2], false, false);
    acc[1][3] = __builtin_amdgcn_wmma_f32_16x16x32_bf16(false, a1, false, b3, (short)0, acc[1][3], false, false);
  }

  // Scatter-store into head-major Q/K and transposed V. Fragment-uniform
  // batch/head indices hoisted; V rows are contiguous -> single b128 store.
#pragma unroll
  for (int mt = 0; mt < 2; ++mt) {
    const int t0 = m0 + mt * 16;                  // fragment's first token
    const int bb = t0 >> 11;
    const int s0 = (t0 & 2047) + 8 * sel;
#pragma unroll
    for (int nt = 0; nt < 4; ++nt) {
      const int n_g   = n0 + nt * 16 + ln;
      const int which = n_g >> 10;                // 0=q 1=k 2=v (tile-uniform)
      const int r     = n_g & 1023;
      const size_t bhq = (size_t)(bb * NH + (r >> 6));
      const int dd    = r & 63;
      if (which == 0) {
        __bf16* p = qb + (bhq * S_LEN + s0) * HD + dd;
#pragma unroll
        for (int j = 0; j < 8; ++j) p[(size_t)j * HD] = (__bf16)(acc[mt][nt][j] * 0.125f);
      } else if (which == 1) {
        __bf16* p = kb + (bhq * S_LEN + s0) * HD + dd;
#pragma unroll
        for (int j = 0; j < 8; ++j) p[(size_t)j * HD] = (__bf16)acc[mt][nt][j];
      } else {
        v8bf pk;
#pragma unroll
        for (int j = 0; j < 8; ++j) pk[j] = (__bf16)acc[mt][nt][j];
        *(v8bf*)(vt + (bhq * HD + dd) * S_LEN + s0) = pk;   // 8 contiguous tokens
      }
    }
  }
}

// ---------------- flash attention ----------------
// grid.x = B*H, grid.y = S/128 ; 8 waves/block, 16 queries per wave.
// K tiles: per-lane async copies (ASYNCcnt). V^T tiles: TDM (TENSORcnt).
__global__ __launch_bounds__(256) void attn_fwd(const __bf16* __restrict__ qb,
                                                const __bf16* __restrict__ kbuf,
                                                const __bf16* __restrict__ vt,
                                                float* __restrict__ out) {
  __shared__ __bf16 Kbuf[2][32 * 64];   // 32 keys x 64 dims, row-major   (8 KB)
  __shared__ __bf16 Vbuf[2][64 * 32];   // 64 dims x 32 keys (V^T tile)   (8 KB)
  __shared__ __bf16 Pbuf[8][16 * 32];   // per-wave P transpose staging   (8 KB)

  const int tid  = threadIdx.x;
  const int lane = tid & 31;
  const int wave = tid >> 5;
  const int sel  = lane >> 4;
  const int ln   = lane & 15;
  const int bh   = blockIdx.x;                    // 0..31
  const int b    = bh >> 4;
  const int h    = bh & 15;
  const int q0b  = blockIdx.y * 128;
  const int q0   = q0b + wave * 16;

  const __bf16* kbase = kbuf + (size_t)bh * S_LEN * HD;
  const __bf16* vbase = vt   + (size_t)bh * HD * S_LEN;
  __bf16* prow = &Pbuf[wave][0];

  // Q fragments: 16 queries x 64 head-dim = 2 A-fragments (K=32 each)
  v16bf qf0, qf1;
  {
    const __bf16* qrp = qb + (size_t)bh * S_LEN * HD + (size_t)(q0 + ln) * HD;
    qf0 = load_frag_a(qrp, sel);
    qf1 = load_frag_a(qrp + 32, sel);
  }

  v8f o[4] = {};
  float mstat[8], lstat[8];
#pragma unroll
  for (int j = 0; j < 8; ++j) { mstat[j] = -__builtin_inff(); lstat[j] = 0.0f; }

  const int nblocks     = (q0b >> 5) + 4;              // key blocks for the WG
  const int kb_end_wave = (((q0 + 15) >> 5) + 1) << 5;

  // prologue: key-block 0 -> buffer 0
  async_cp16(kbase + (size_t)tid * 8, &Kbuf[0][tid * 8]);   // 256 thr x 16B = 4KB
  if (wave == 0) tdm_load_vtile(vbase, &Vbuf[0][0]);        // one TDM op

  for (int i = 0; i < nblocks; ++i) {
    const int cur = i & 1;
    if (i + 1 < nblocks) {
      const int nb = (i + 1) & 1;
      const int kn = (i + 1) * 32;
      async_cp16(kbase + (size_t)kn * HD + tid * 8, &Kbuf[nb][tid * 8]);
      if (wave == 0) tdm_load_vtile(vbase + kn, &Vbuf[nb][0]);
      asm volatile("s_wait_asynccnt 0x1" ::: "memory");      // K tile i landed
      if (wave == 0) asm volatile("s_wait_tensorcnt 0x1" ::: "memory");
    } else {
      asm volatile("s_wait_asynccnt 0x0" ::: "memory");
      if (wave == 0) asm volatile("s_wait_tensorcnt 0x0" ::: "memory");
    }
    __syncthreads();                                         // tile i visible

    const int kblk = i * 32;
    if (kblk < kb_end_wave) {                 // wave-uniform causal skip
      // ---- S = Q K^T (two 16-key tiles, K-dim = 64 as 2x32), K from LDS ----
      v8f sc[2] = {};
#pragma unroll
      for (int t = 0; t < 2; ++t) {
        const __bf16* kcol = &Kbuf[cur][(t * 16 + ln) * HD];
        v16bf b0 = load_frag_b(kcol, sel);
        v16bf b1 = load_frag_b(kcol + 32, sel);
        sc[t] = __builtin_amdgcn_wmma_f32_16x16x32_bf16(false, qf0, false, b0, (short)0, sc[t], false, false);
        sc[t] = __builtin_amdgcn_wmma_f32_16x16x32_bf16(false, qf1, false, b1, (short)0, sc[t], false, false);
      }

      // ---- causal mask ----
#pragma unroll
      for (int t = 0; t < 2; ++t) {
        const int kj = kblk + t * 16 + ln;
#pragma unroll
        for (int j = 0; j < 8; ++j) {
          const int qi = q0 + j + 8 * sel;
          if (kj > qi) sc[t][j] = -__builtin_inff();
        }
      }

      // ---- online softmax (row stats per lane-half, replicated) ----
#pragma unroll
      for (int j = 0; j < 8; ++j) {
        float rmax = half_red_max(fmaxf(sc[0][j], sc[1][j]));
        const float mnew  = fmaxf(mstat[j], rmax);
        const float alpha = __expf(mstat[j] - mnew);
        const float p0 = __expf(sc[0][j] - mnew);
        const float p1 = __expf(sc[1][j] - mnew);
        sc[0][j] = p0; sc[1][j] = p1;
        lstat[j] = lstat[j] * alpha + half_red_sum(p0 + p1);
        mstat[j] = mnew;
#pragma unroll
        for (int c = 0; c < 4; ++c) o[c][j] *= alpha;
      }

      // ---- transpose P (C-layout f32) -> A-fragment (bf16) via per-wave LDS ----
#pragma unroll
      for (int t = 0; t < 2; ++t)
#pragma unroll
        for (int j = 0; j < 8; ++j)
          prow[(j + 8 * sel) * 32 + t * 16 + ln] = (__bf16)sc[t][j];
      asm volatile("s_wait_dscnt 0" ::: "memory");     // same-wave LDS ordering
      v16bf pf = load_frag_a(prow + ln * 32, sel);

      // ---- O += P @ V^T (K-dim = 32 keys), V from LDS ----
#pragma unroll
      for (int c = 0; c < 4; ++c) {
        const __bf16* vcol = &Vbuf[cur][(c * 16 + ln) * 32];
        v16bf vf = load_frag_b(vcol, sel);
        o[c] = __builtin_amdgcn_wmma_f32_16x16x32_bf16(false, pf, false, vf, (short)0, o[c], false, false);
      }
    }
    __syncthreads();      // all readers done with buf[cur] before reuse
  }

  // ---- epilogue: O / l ----
  float inv_l[8];
#pragma unroll
  for (int j = 0; j < 8; ++j) inv_l[j] = 1.0f / lstat[j];
  float* obase = out + (size_t)b * S_LEN * DMODEL + h * HD;
#pragma unroll
  for (int c = 0; c < 4; ++c) {
    const int dd = c * 16 + ln;
#pragma unroll
    for (int j = 0; j < 8; ++j) {
      const int srow = q0 + j + 8 * sel;
      obase[(size_t)srow * DMODEL + dd] = o[c][j] * inv_l[j];
    }
  }
}

// ---------------- launcher ----------------
extern "C" void kernel_launch(void* const* d_in, const int* in_sizes, int n_in,
                              void* d_out, int out_size, void* d_ws, size_t ws_size,
                              hipStream_t stream) {
  (void)in_sizes; (void)n_in; (void)out_size; (void)ws_size;
  const float* x     = (const float*)d_in[0];   // [2,2048,1024]
  const float* w_qkv = (const float*)d_in[1];   // [1024,3072]
  // d_in[2] = w_o is unused by the reference forward.
  float* out = (float*)d_out;                   // [2,2048,1024] fp32

  const size_t XB_E  = (size_t)4096 * 1024;
  const size_t WT_E  = (size_t)3072 * 1024;
  const size_t QKV_E = (size_t)BHCNT * S_LEN * HD;

  __bf16* xb = (__bf16*)d_ws;
  __bf16* wt = xb + XB_E;
  __bf16* qb = wt + WT_E;
  __bf16* kb = qb + QKV_E;
  __bf16* vt = kb + QKV_E;   // ~38 MB of workspace total

  cvt_x_bf16<<<dim3((unsigned)(XB_E / 4 / 256)), dim3(256), 0, stream>>>((const float4*)x, xb);
  cvt_wt_bf16<<<dim3((unsigned)(WT_E / 256)), dim3(256), 0, stream>>>(w_qkv, wt);
  qkv_gemm<<<dim3(3072 / 128, 4096 / 128), dim3(256), 0, stream>>>(xb, wt, qb, kb, vt);
  attn_fwd<<<dim3(BHCNT, S_LEN / 128), dim3(256), 0, stream>>>(qb, kb, vt, out);
}
